// HybridAttention_36129264894200
// MI455X (gfx1250) — matrix operand order repaired and checked by reading, not compiled, not run on hardware
//
#include <hip/hip_runtime.h>
#include <hip/hip_bf16.h>
#include <math.h>

// ---------------------------------------------------------------------------
// HybridAttention for MI455X (gfx1250, wave32, WMMA).
// GEMMs on v_wmma_f32_16x16x32_bf16 with 2x4 register tiling per wave
// (32x64 strip, 8 accumulators); bf16 intermediates kept L2-resident;
// only attns/out touch HBM in f32.
// ---------------------------------------------------------------------------

typedef __attribute__((ext_vector_type(16))) __bf16 v16bf;
typedef __attribute__((ext_vector_type(8)))  float  v8f;

constexpr int NB = 8, NT = 512, NL = 2048, NC = 256;

// d_out tuple layout: out (B,C,L) | attns (B,2,L,T) | real_sigma (2)
constexpr long ATTN_OFF = (long)NB * NC * NL;                 // 4,194,304
constexpr long SIG_OFF  = ATTN_OFF + (long)NB * 2 * NL * NT;  // 20,971,520

// workspace layout (bytes)
constexpr size_t WS_H    = 0;                                      // bf16 h    [B][2][T][C]
constexpr size_t WS_ATTN = WS_H    + (size_t)NB * 2 * NT * NC * 2; // bf16 attn [B][2][L][T]
constexpr size_t WS_OUT1 = WS_ATTN + (size_t)NB * 2 * NL * NT * 2; // bf16 out1T[B][2C][L]

// ---------------------------------------------------------------------------
// One wave computes a (16*MT) x (16*NTN) strip of D = A((16*MT)xK) * B(Kx16*NTN).
// A fragment (16-bit 16x32 per step, ISA 7.12.2): lane m = lane&15,
//   half = lane>>4, element e holds K = (e>>3)*16 + half*8 + (e&7).
// B fragment (32x16 per step): lane n = lane&15, element e holds the same K as A.
// A fragments loaded once per K-step, B fragments reused across MT rows:
// MT*NTN independent WMMAs per K-step cover the XDL hazard window.
// Sources may be f32 (converted to bf16 in-register) or bf16.
// ---------------------------------------------------------------------------
template <int MT, int NTN, typename TA, typename TB>
__device__ __forceinline__ void wave_gemm(const TA* __restrict__ A, long rsA, long csA,
                                          const TB* __restrict__ B, long rsB, long csB,
                                          int K, v8f acc[MT][NTN])
{
    const int lane = threadIdx.x & 31;
    const int hf   = lane >> 4;
    const int m    = lane & 15;
    for (int k0 = 0; k0 < K; k0 += 32) {
        // speculative prefetch of the next K-chunk (dropped if past the end)
        __builtin_prefetch((const void*)(A + (long)m * rsA + (long)(k0 + 32) * csA), 0, 3);
        __builtin_prefetch((const void*)(B + (long)(k0 + 32 + lane) * rsB), 0, 3);

        v16bf afrag[MT];
#pragma unroll
        for (int mt = 0; mt < MT; ++mt)
#pragma unroll
            for (int e = 0; e < 16; ++e) {
                const int kk = ((e >> 3) << 4) + (hf << 3) + (e & 7);
                afrag[mt][e] = (__bf16)A[(long)(mt * 16 + m) * rsA + (long)(k0 + kk) * csA];
            }
#pragma unroll
        for (int nt = 0; nt < NTN; ++nt) {
            v16bf bfrag;
#pragma unroll
            for (int e = 0; e < 16; ++e) {
                const int kk = ((e >> 3) << 4) + (hf << 3) + (e & 7);
                bfrag[e] = (__bf16)B[(long)(k0 + kk) * rsB + (long)(nt * 16 + m) * csB];
            }
#pragma unroll
            for (int mt = 0; mt < MT; ++mt)
                acc[mt][nt] = __builtin_amdgcn_wmma_f32_16x16x32_bf16(
                    /*neg_a=*/false, afrag[mt], /*neg_b=*/false, bfrag,
                    /*c_mod=*/(short)0, acc[mt][nt], /*reuse_a=*/false, /*reuse_b=*/false);
        }
    }
}

// ---------------------------------------------------------------------------
// h[b][g][t][c] = sum_k w_hidden[g*C+c][k] * x_h[b][k][t] + b_hidden[g*C+c]
// Wave strip: 16 t-rows x 64 c-cols. 128 strips per (b,g), 16 (b,g) pairs.
// ---------------------------------------------------------------------------
__global__ void k_hidden(const float* __restrict__ x_h,
                         const float* __restrict__ w_hidden,
                         const float* __restrict__ b_hidden,
                         __bf16* __restrict__ h_bf)
{
    const int wav = (blockIdx.x * blockDim.x + threadIdx.x) >> 5;
    const int bg  = wav >> 7;            // 128 strips per bg
    const int r   = wav & 127;
    const int t0  = (r >> 2) << 4;       // 32 t-tiles
    const int c0  = (r & 3) << 6;        // 4 c-strips of 64
    const int b   = bg >> 1, g = bg & 1;

    // A[t][k] = x_h[b][k][t]       -> rs(t)=1, cs(k)=T   (f32, converted)
    const float* A  = x_h      + (long)b * NC * NT + t0;
    // B[k][c] = w_hidden[g*C+c][k] -> rs(k)=1, cs(c)=C   (f32, converted)
    const float* Bm = w_hidden + (long)(g * NC + c0) * NC;
    v8f acc[1][4] = {};
    wave_gemm<1, 4>(A, 1L, (long)NT, Bm, 1L, (long)NC, NC, acc);

    const int lane = threadIdx.x & 31;
    const int hf = lane >> 4, n = lane & 15;
    __bf16* dst = h_bf + (long)bg * NT * NC;
#pragma unroll
    for (int nt = 0; nt < 4; ++nt) {
        const int cc = c0 + nt * 16 + n;
        const float bias = b_hidden[g * NC + cc];
#pragma unroll
        for (int v = 0; v < 8; ++v) {
            const int mr = v + (hf << 3);
            dst[(long)(t0 + mr) * NC + cc] = (__bf16)(acc[0][nt][v] + bias);
        }
    }
}

// ---------------------------------------------------------------------------
// Softmax over T=512 per (b,s,l) row: one wave32 per row, 16 elems/lane.
// Writes f32 attns to d_out and a bf16 copy to workspace for the WMMA GEMM.
// ---------------------------------------------------------------------------
__global__ void k_softmax(const float* __restrict__ e,
                          const float* __restrict__ a,
                          const float* __restrict__ bnd,
                          const unsigned char* __restrict__ tmask,
                          const float* __restrict__ mmask,
                          const float* __restrict__ sigma,
                          float* __restrict__ attn_f32,
                          __bf16* __restrict__ attn_bf)
{
    const int w    = (blockIdx.x * blockDim.x + threadIdx.x) >> 5;  // row id
    const int lane = threadIdx.x & 31;
    const int b = w >> 12;          // / (2*L)
    const int s = (w >> 11) & 1;    // / L mod 2
    const int l = w & (NL - 1);

    const float sg = fminf(fmaxf(sigma[s], 1e-6f), 3.0f);
    const float q  = (float)l * mmask[(long)b * NL + l];

    float ev[16];
    float mx = -INFINITY;
#pragma unroll
    for (int i = 0; i < 16; ++i) {
        const int t = lane + (i << 5);
        float en;
        if (s == 0) {
            const float d = q - e[(long)b * NT + t];
            en = -d * d;
        } else {
            const float av = a[(long)b * NT + t];
            const float bv = bnd[(long)b * NT + t];
            const float d  = fabsf(q - av) + fabsf(q - bv) - (bv - av);
            en = -d * d;
        }
        en *= sg;
        if (!tmask[(long)b * NT + t]) en = -INFINITY;
        ev[i] = en;
        mx = fmaxf(mx, en);
    }
#pragma unroll
    for (int off = 16; off >= 1; off >>= 1)
        mx = fmaxf(mx, __shfl_xor(mx, off, 32));

    float sum = 0.f;
#pragma unroll
    for (int i = 0; i < 16; ++i) {
        const float ex = __expf(ev[i] - mx);
        ev[i] = ex;
        sum += ex;
    }
#pragma unroll
    for (int off = 16; off >= 1; off >>= 1)
        sum += __shfl_xor(sum, off, 32);

    const float inv = 1.0f / sum;
    const long base = (long)w * NT;
#pragma unroll
    for (int i = 0; i < 16; ++i) {
        const int t = lane + (i << 5);
        const float p = ev[i] * inv;
        attn_f32[base + t] = p;
        attn_bf[base + t]  = (__bf16)p;
    }
}

// ---------------------------------------------------------------------------
// out1T[b][g*C+c][l] = sum_t attns[b][g][l][t] * h[b][g][t][c]
// Dominant GEMM: per (b,g) 2048x512 * 512x256.
// Wave strip: 32 l-rows x 64 c-cols -> 256 strips per (b,g).
// Transposed store: per lane, 8 consecutive l make 16B contiguous stores.
// ---------------------------------------------------------------------------
__global__ void k_ctx(const __bf16* __restrict__ attn_bf,
                      const __bf16* __restrict__ h_bf,
                      __bf16* __restrict__ out1T)
{
    const int wav = (blockIdx.x * blockDim.x + threadIdx.x) >> 5;
    const int bg  = wav >> 8;            // 256 strips per bg
    const int r   = wav & 255;
    const int l0  = (r >> 2) << 5;       // 64 l-strips of 32
    const int c0  = (r & 3) << 6;        // 4 c-strips of 64
    const int b   = bg >> 1, g = bg & 1;

    const __bf16* A  = attn_bf + ((long)bg * NL + l0) * NT;   // rs=T, cs=1
    const __bf16* Bm = h_bf    + (long)bg * NT * NC + c0;     // rs=C, cs=1
    v8f acc[2][4] = {};
    wave_gemm<2, 4>(A, (long)NT, 1L, Bm, (long)NC, 1L, NT, acc);

    const int lane = threadIdx.x & 31;
    const int hf = lane >> 4, n = lane & 15;
#pragma unroll
    for (int nt = 0; nt < 4; ++nt) {
        const int cc = g * NC + c0 + nt * 16 + n;
        __bf16* dst = out1T + ((long)b * (2 * NC) + cc) * NL + l0;
#pragma unroll
        for (int mt = 0; mt < 2; ++mt)
#pragma unroll
            for (int v = 0; v < 8; ++v) {
                const int mr = mt * 16 + (hf << 3) + v;   // consecutive in v
                dst[mr] = (__bf16)acc[mt][nt][v];
            }
    }
}

// ---------------------------------------------------------------------------
// out[b][o][l] = b_out[o] + sum_k w_out[o][k] * out1T[b][k][l]   (K = 2C = 512)
// Wave strip: 32 o-rows x 64 l-cols -> 256 strips per batch.
// B-fragment: per element, 16 lanes read 32 contiguous bytes (16 l) of out1T.
// ---------------------------------------------------------------------------
__global__ void k_out(const float* __restrict__ w_out,
                      const float* __restrict__ b_out,
                      const __bf16* __restrict__ out1T,
                      float* __restrict__ out)
{
    const int wav = (blockIdx.x * blockDim.x + threadIdx.x) >> 5;
    const int b   = wav >> 8;            // 256 strips per batch
    const int r   = wav & 255;
    const int o0  = (r >> 5) << 5;       // 8 o-strips of 32
    const int l0  = (r & 31) << 6;       // 32 l-strips of 64

    const float*  A  = w_out + (long)o0 * (2 * NC);            // rs=512, cs=1 (f32)
    const __bf16* Bm = out1T + (long)b * (2 * NC) * NL + l0;   // B[k][l]: rs=NL, cs=1
    v8f acc[2][4] = {};
    wave_gemm<2, 4>(A, (long)(2 * NC), 1L, Bm, (long)NL, 1L, 2 * NC, acc);

    const int lane = threadIdx.x & 31;
    const int hf = lane >> 4, n = lane & 15;
#pragma unroll
    for (int nt = 0; nt < 4; ++nt) {
        const int ll = l0 + nt * 16 + n;
#pragma unroll
        for (int mt = 0; mt < 2; ++mt)
#pragma unroll
            for (int v = 0; v < 8; ++v) {
                const int mr = o0 + mt * 16 + (hf << 3) + v;
                out[((long)b * NC + mr) * NL + ll] = acc[mt][nt][v] + b_out[mr];
            }
    }
}

__global__ void k_sigma(const float* __restrict__ sigma, float* __restrict__ out_sig)
{
    if (threadIdx.x < 2)
        out_sig[threadIdx.x] = fminf(fmaxf(sigma[threadIdx.x], 1e-6f), 3.0f);
}

// ---------------------------------------------------------------------------
extern "C" void kernel_launch(void* const* d_in, const int* in_sizes, int n_in,
                              void* d_out, int out_size, void* d_ws, size_t ws_size,
                              hipStream_t stream)
{
    const float*         e        = (const float*)d_in[0];
    const float*         a        = (const float*)d_in[1];
    const float*         bnd      = (const float*)d_in[2];
    const float*         x_h      = (const float*)d_in[3];
    const unsigned char* tmask    = (const unsigned char*)d_in[4];
    const float*         mmask    = (const float*)d_in[5];
    const float*         sigma    = (const float*)d_in[6];
    const float*         w_hidden = (const float*)d_in[7];
    const float*         b_hidden = (const float*)d_in[8];
    const float*         w_out    = (const float*)d_in[9];
    const float*         b_out    = (const float*)d_in[10];

    float* out = (float*)d_out;
    char*  ws  = (char*)d_ws;
    __bf16* h_bf    = (__bf16*)(ws + WS_H);
    __bf16* attn_bf = (__bf16*)(ws + WS_ATTN);
    __bf16* out1T   = (__bf16*)(ws + WS_OUT1);

    // 8 waves / block (256 threads); one wave per strip (or softmax row).
    k_hidden <<< (16 * 128) / 8, 256, 0, stream >>>(x_h, w_hidden, b_hidden, h_bf);
    k_softmax<<< (NB * 2 * NL) / 8, 256, 0, stream >>>(e, a, bnd, tmask, mmask, sigma,
                                                       out + ATTN_OFF, attn_bf);
    k_sigma  <<< 1, 32, 0, stream >>>(sigma, out + SIG_OFF);
    k_ctx    <<< (16 * 256) / 8, 256, 0, stream >>>(attn_bf, h_bf, out1T);
    k_out    <<< (NB * 256) / 8, 256, 0, stream >>>(w_out, b_out, out1T, out);
}
